// Mamba_block_88476326298069
// MI455X (gfx1250) — matrix operand order repaired and checked by reading, not compiled, not run on hardware
//
#include <hip/hip_runtime.h>
#include <hip/hip_bf16.h>

typedef __attribute__((ext_vector_type(16))) _Float16 v16h;
typedef __attribute__((ext_vector_type(8)))  _Float16 v8h;
typedef __attribute__((ext_vector_type(8)))  float    v8f;

#define BB   4
#define CC   512
#define LL   1024
#define DI   1024      // d_inner
#define XZW  2048      // 2*d_inner
#define DTR  32        // dt_rank
#define NST  16        // D_STATE
#define MM   (BB*LL)   // 4096 rows

// ---------------------------------------------------------------------------
// Weight convert + transpose: Wt[n*K + k] = (f16) W[k*N + n]
// ---------------------------------------------------------------------------
__global__ void wt_transpose(const float* __restrict__ W, _Float16* __restrict__ Wt,
                             int K, int N) {
    int idx = blockIdx.x * blockDim.x + threadIdx.x;
    if (idx >= K * N) return;
    int n = idx / K;
    int k = idx - n * K;
    Wt[idx] = (_Float16)W[(size_t)k * N + n];
}

// ---------------------------------------------------------------------------
// LayerNorm over C=512 per (b,l); emits f16 A-matrix rows (M x C).
// features layout (B,C,H,W): x[b,l,c] = features[b*C*L + c*L + l]
// ---------------------------------------------------------------------------
__global__ void ln_kernel(const float* __restrict__ f,
                          const float* __restrict__ w,
                          const float* __restrict__ bbias,
                          _Float16* __restrict__ xn16) {
    int bl = blockIdx.x;            // 0..4095
    int b  = bl >> 10;
    int l  = bl & 1023;
    int t  = threadIdx.x;           // 256 threads, 2 channels each
    const float* base = f + ((size_t)b * CC) * LL + l;
    float v0 = base[(size_t)t * LL];
    float v1 = base[(size_t)(t + 256) * LL];

    __shared__ float s[256];
    s[t] = v0 + v1;
    __syncthreads();
    #pragma unroll
    for (int o = 128; o > 0; o >>= 1) {
        if (t < o) s[t] += s[t + o];
        __syncthreads();
    }
    float mu = s[0] * (1.0f / CC);
    __syncthreads();

    float d0 = v0 - mu, d1 = v1 - mu;
    s[t] = d0 * d0 + d1 * d1;
    __syncthreads();
    #pragma unroll
    for (int o = 128; o > 0; o >>= 1) {
        if (t < o) s[t] += s[t + o];
        __syncthreads();
    }
    float rstd = rsqrtf(s[0] * (1.0f / CC) + 1e-5f);

    _Float16* o16 = xn16 + (size_t)bl * CC;
    o16[t]       = (_Float16)(d0 * rstd * w[t]       + bbias[t]);
    o16[t + 256] = (_Float16)(d1 * rstd * w[t + 256] + bbias[t + 256]);
}

// ---------------------------------------------------------------------------
// WMMA GEMM: C(MxN,f32) = A(MxK,f16,row-major) * Bt(NxK,f16,row-major)^T
// Block = 128 threads = 4 waves in 2x2; each wave computes a 32x32 output
// patch as a 2x2 grid of 16x16 WMMA tiles. K stepped by 32.
// N, K are compile-time so store/load offsets fold into immediates.
// M,N multiples of 64; K multiple of 32.
// ---------------------------------------------------------------------------
template <int N, int K>
__global__ __launch_bounds__(128)
void wmma_gemm(const _Float16* __restrict__ A,
               const _Float16* __restrict__ Bt,
               float* __restrict__ Cm) {
    const int lane = threadIdx.x & 31;
    const int wave = threadIdx.x >> 5;
    const int m16  = lane & 15;
    const int half = lane >> 4;
    const int wm   = wave >> 1;
    const int wn   = wave & 1;
    const int blockM = blockIdx.y * 64 + wm * 32;
    const int blockN = blockIdx.x * 64 + wn * 32;

    v8f acc[2][2] = {};

    // per-lane fragment base pointers (advance by 32 f16 per k-step)
    const _Float16* ap0 = A  + (size_t)(blockM + m16) * K      + half * 8;
    const _Float16* ap1 = ap0 + (size_t)16 * K;
    const _Float16* bp0 = Bt + (size_t)(blockN + m16) * K      + half * 8;
    const _Float16* bp1 = bp0 + (size_t)16 * K;

    for (int k0 = 0; k0 < K; k0 += 32) {
        v16h a[2], bfrag[2];
        {
            v8h lo = *(const v8h*)ap0;
            v8h hi = *(const v8h*)(ap0 + 16);
            a[0] = __builtin_shufflevector(lo, hi, 0,1,2,3,4,5,6,7,8,9,10,11,12,13,14,15);
            lo = *(const v8h*)ap1;
            hi = *(const v8h*)(ap1 + 16);
            a[1] = __builtin_shufflevector(lo, hi, 0,1,2,3,4,5,6,7,8,9,10,11,12,13,14,15);
            v8h blo = *(const v8h*)bp0;
            v8h bhi = *(const v8h*)(bp0 + 16);
            bfrag[0] = __builtin_shufflevector(blo, bhi, 0,1,2,3,4,5,6,7,8,9,10,11,12,13,14,15);
            blo = *(const v8h*)bp1;
            bhi = *(const v8h*)(bp1 + 16);
            bfrag[1] = __builtin_shufflevector(blo, bhi, 0,1,2,3,4,5,6,7,8,9,10,11,12,13,14,15);
        }
        ap0 += 32; ap1 += 32; bp0 += 32; bp1 += 32;

        #pragma unroll
        for (int i = 0; i < 2; ++i)
            #pragma unroll
            for (int j = 0; j < 2; ++j)
                acc[i][j] = __builtin_amdgcn_wmma_f32_16x16x32_f16(
                    false, a[i], false, bfrag[j], (short)0, acc[i][j], false, false);
    }

    // D layout: VGPR r, lane half h -> row = r + 8h, col = lane&15.
    // One base address per i-tile; all other offsets are compile-time
    // immediates (r*N*4 + j*64 bytes).
    #pragma unroll
    for (int i = 0; i < 2; ++i) {
        float* base = Cm + (size_t)(blockM + i * 16 + half * 8) * N + blockN + m16;
        #pragma unroll
        for (int j = 0; j < 2; ++j)
            #pragma unroll
            for (int r = 0; r < 8; ++r)
                base[r * N + j * 16] = acc[i][j][r];
    }
}

// ---------------------------------------------------------------------------
// Depthwise causal conv (k=4) + bias + SiLU. xm = xz[:, :DI].
// Writes xs fp32 and xs f16 (next GEMM's A).
// ---------------------------------------------------------------------------
__global__ void conv_silu(const float* __restrict__ xz,
                          const float* __restrict__ cw,
                          const float* __restrict__ cb,
                          float* __restrict__ xs,
                          _Float16* __restrict__ xs16) {
    int idx = blockIdx.x * blockDim.x + threadIdx.x;   // M*DI
    if (idx >= MM * DI) return;
    int d  = idx & (DI - 1);
    int bl = idx >> 10;           // DI = 1024
    int l  = bl & (LL - 1);
    int b  = bl >> 10;
    float acc = cb[d];
    #pragma unroll
    for (int k = 0; k < 4; ++k) {
        int ls = l - 3 + k;
        if (ls >= 0)
            acc += cw[d * 4 + k] * xz[((size_t)(b * LL + ls)) * XZW + d];
    }
    float sv = acc / (1.0f + __expf(-acc));   // SiLU
    xs[idx]   = sv;
    xs16[idx] = (_Float16)sv;
}

// proj[:, :32] -> f16 A-matrix (M x 32)
__global__ void projdt_f16(const float* __restrict__ proj, _Float16* __restrict__ a16) {
    int idx = blockIdx.x * blockDim.x + threadIdx.x;   // M*32
    if (idx >= MM * DTR) return;
    int m = idx >> 5;
    int c = idx & 31;
    a16[idx] = (_Float16)proj[(size_t)m * 64 + c];
}

// dt = softplus(dtlin + b_dt), in place
__global__ void dt_softplus(float* __restrict__ dt, const float* __restrict__ bdt) {
    int idx = blockIdx.x * blockDim.x + threadIdx.x;   // M*DI
    if (idx >= MM * DI) return;
    int d = idx & (DI - 1);
    float v = dt[idx] + bdt[d];
    dt[idx] = (v > 20.0f) ? v : log1pf(__expf(v));
}

// ---------------------------------------------------------------------------
// Selective scan: one thread per (b,d); h[16] in registers; 1024 serial steps.
// B_t = proj[m*64 + 32 + s], C_t = proj[m*64 + 48 + s] (uniform per wave).
// ---------------------------------------------------------------------------
__global__ void scan_kernel(const float* __restrict__ dt,
                            const float* __restrict__ proj,
                            const float* __restrict__ xs,
                            const float* __restrict__ A_log,
                            float* __restrict__ ys) {
    int tid = blockIdx.x * blockDim.x + threadIdx.x;   // B*DI = 4096
    if (tid >= BB * DI) return;
    int b = tid >> 10;
    int d = tid & (DI - 1);

    float A[NST];
    #pragma unroll
    for (int s = 0; s < NST; ++s) A[s] = -__expf(A_log[d * NST + s]);

    float h[NST];
    #pragma unroll
    for (int s = 0; s < NST; ++s) h[s] = 0.0f;

    for (int l = 0; l < LL; ++l) {
        size_t m = (size_t)b * LL + l;
        float dtv = dt[m * DI + d];
        float xv  = xs[m * DI + d];
        float dtx = dtv * xv;
        const float* pr = proj + m * 64;
        float y = 0.0f;
        #pragma unroll
        for (int s = 0; s < NST; ++s) {
            float dA = __expf(dtv * A[s]);
            h[s] = dA * h[s] + dtx * pr[32 + s];
            y   += h[s] * pr[48 + s];
        }
        ys[m * DI + d] = y;
    }
}

// y = (ys + xs*D) * silu(z); emit f16 A-matrix for final GEMM
__global__ void gate_kernel(const float* __restrict__ ys,
                            const float* __restrict__ xs,
                            const float* __restrict__ xz,
                            const float* __restrict__ Dp,
                            _Float16* __restrict__ y16) {
    int idx = blockIdx.x * blockDim.x + threadIdx.x;   // M*DI
    if (idx >= MM * DI) return;
    int d = idx & (DI - 1);
    int m = idx >> 10;
    float z = xz[(size_t)m * XZW + DI + d];
    float v = (ys[idx] + xs[idx] * Dp[d]) * (z / (1.0f + __expf(-z)));
    y16[idx] = (_Float16)v;
}

// (B*L, C) fp32 -> (B, C, H, W) fp32
__global__ void final_transpose(const float* __restrict__ op, float* __restrict__ out) {
    int idx = blockIdx.x * blockDim.x + threadIdx.x;   // B*C*L
    if (idx >= BB * CC * LL) return;
    int l = idx & (LL - 1);
    int c = (idx >> 10) & (CC - 1);
    int b = idx >> 19;
    out[idx] = op[((size_t)b * LL + l) * CC + c];
}

// ---------------------------------------------------------------------------
extern "C" void kernel_launch(void* const* d_in, const int* in_sizes, int n_in,
                              void* d_out, int out_size, void* d_ws, size_t ws_size,
                              hipStream_t stream) {
    const float* features = (const float*)d_in[0];
    const float* ln_w     = (const float*)d_in[1];
    const float* ln_b     = (const float*)d_in[2];
    const float* W_in     = (const float*)d_in[3];   // (512, 2048)
    const float* conv_w   = (const float*)d_in[4];   // (1024, 4)
    const float* conv_b   = (const float*)d_in[5];   // (1024,)
    const float* W_x      = (const float*)d_in[6];   // (1024, 64)
    const float* W_dt     = (const float*)d_in[7];   // (32, 1024)
    const float* b_dt     = (const float*)d_in[8];   // (1024,)
    const float* A_log    = (const float*)d_in[9];   // (1024, 16)
    const float* D_param  = (const float*)d_in[10];  // (1024,)
    const float* W_out    = (const float*)d_in[11];  // (1024, 512)
    float* out = (float*)d_out;

    char* wp = (char*)d_ws;
    auto carve = [&](size_t bytes) -> void* {
        void* p = (void*)wp;
        wp += (bytes + 255) & ~(size_t)255;
        return p;
    };
    _Float16* xn16   = (_Float16*)carve((size_t)MM * CC * 2);
    _Float16* Wt_in  = (_Float16*)carve((size_t)XZW * CC * 2);
    _Float16* Wt_x   = (_Float16*)carve((size_t)64 * DI * 2);
    _Float16* Wt_dt  = (_Float16*)carve((size_t)DI * DTR * 2);
    _Float16* Wt_out = (_Float16*)carve((size_t)CC * DI * 2);
    float*    xz     = (float*)   carve((size_t)MM * XZW * 4);
    float*    xs     = (float*)   carve((size_t)MM * DI * 4);
    _Float16* xs16   = (_Float16*)carve((size_t)MM * DI * 2);
    float*    proj   = (float*)   carve((size_t)MM * 64 * 4);
    _Float16* pdt16  = (_Float16*)carve((size_t)MM * DTR * 2);
    float*    dtbuf  = (float*)   carve((size_t)MM * DI * 4);  // dtlin -> dt (in place)
    float*    ysbuf  = (float*)   carve((size_t)MM * DI * 4);
    _Float16* y16    = (_Float16*)carve((size_t)MM * DI * 2);
    float*    outp   = (float*)   carve((size_t)MM * CC * 4);

    const int T = 256;

    // weight conversion/transpose
    wt_transpose<<<(CC * XZW + T - 1) / T, T, 0, stream>>>(W_in,  Wt_in,  CC, XZW);
    wt_transpose<<<(DI * 64  + T - 1) / T, T, 0, stream>>>(W_x,   Wt_x,   DI, 64);
    wt_transpose<<<(DTR * DI + T - 1) / T, T, 0, stream>>>(W_dt,  Wt_dt,  DTR, DI);
    wt_transpose<<<(DI * CC  + T - 1) / T, T, 0, stream>>>(W_out, Wt_out, DI, CC);

    // LayerNorm -> f16 rows
    ln_kernel<<<MM, 256, 0, stream>>>(features, ln_w, ln_b, xn16);

    // xz = xn @ W_in   (4096 x 2048 x 512)
    wmma_gemm<XZW, CC><<<dim3(XZW / 64, MM / 64), 128, 0, stream>>>(xn16, Wt_in, xz);

    // causal depthwise conv + SiLU
    conv_silu<<<(MM * DI) / T, T, 0, stream>>>(xz, conv_w, conv_b, xs, xs16);

    // proj = xs @ W_x  (4096 x 64 x 1024)
    wmma_gemm<64, DI><<<dim3(64 / 64, MM / 64), 128, 0, stream>>>(xs16, Wt_x, proj);

    // dt path
    projdt_f16<<<(MM * DTR) / T, T, 0, stream>>>(proj, pdt16);
    wmma_gemm<DI, DTR><<<dim3(DI / 64, MM / 64), 128, 0, stream>>>(pdt16, Wt_dt, dtbuf);
    dt_softplus<<<(MM * DI) / T, T, 0, stream>>>(dtbuf, b_dt);

    // selective scan
    scan_kernel<<<(BB * DI) / T, T, 0, stream>>>(dtbuf, proj, xs, A_log, ysbuf);

    // gating + f16 conversion
    gate_kernel<<<(MM * DI) / T, T, 0, stream>>>(ysbuf, xs, xz, D_param, y16);

    // out = y @ W_out  (4096 x 512 x 1024)
    wmma_gemm<CC, DI><<<dim3(CC / 64, MM / 64), 128, 0, stream>>>(y16, Wt_out, outp);

    // layout to (B, C, H, W)
    final_transpose<<<(BB * CC * LL) / T, T, 0, stream>>>(outp, out);
}
